// GINConv_88244398064422
// MI455X (gfx1250) — matrix operand order repaired and checked by reading, not compiled, not run on hardware
//
#include <hip/hip_runtime.h>

// ---------------------------------------------------------------------------
// GINConv fused pipeline for MI455X (gfx1250, wave32, WMMA)
//   deg/agg scatter (fp32 atomics)  ->  bf16 concat  ->  WMMA GEMM1+ReLU+BN
//   partials -> BN fold into W2     ->  WMMA GEMM2 -> fp32 out
// ---------------------------------------------------------------------------

typedef __bf16 bf16_t;
typedef __attribute__((ext_vector_type(16))) __bf16 v16bf;
typedef __attribute__((ext_vector_type(8)))  __bf16 v8bf;
typedef __attribute__((ext_vector_type(8)))  float  v8f;

__device__ __forceinline__ bf16_t f2bf(float f) {
  // round-to-nearest-even fp32 -> bf16
  unsigned u = __builtin_bit_cast(unsigned, f);
  u += 0x7FFFu + ((u >> 16) & 1u);
  unsigned short h = (unsigned short)(u >> 16);
  return __builtin_bit_cast(bf16_t, h);
}

// ---------------------------------------------------------------------------
__global__ __launch_bounds__(256) void zero_f32(float* __restrict__ p, size_t n) {
  size_t i = (size_t)blockIdx.x * 256 + threadIdx.x;
  if (i < n) p[i] = 0.f;
}

// One wave (32 lanes) per edge: gather x[col], scale by vals, atomic scatter
// into agg[row] (192 feats: 128 x_obj + 64 x_e) + deg accumulation.
__global__ __launch_bounds__(256) void edge_scatter(
    const float* __restrict__ x, const float* __restrict__ vals,
    const float* __restrict__ ea, const int* __restrict__ row,
    const int* __restrict__ col, float* __restrict__ deg,
    float* __restrict__ agg, int E) {
  int e = (int)(((unsigned)blockIdx.x * 256u + threadIdx.x) >> 5);
  int lane = threadIdx.x & 31;
  if (e >= E) return;
  int r = row[e];
  int c = col[e];
  float v = vals[e];
  const float* xs = x + (size_t)c * 128;
  float* ar = agg + (size_t)r * 192;
#pragma unroll
  for (int i = 0; i < 4; ++i) {
    float pv = v * xs[lane + 32 * i];
    __hip_atomic_fetch_add(&ar[lane + 32 * i], pv, __ATOMIC_RELAXED,
                           __HIP_MEMORY_SCOPE_AGENT);
  }
  const float* es = ea + (size_t)e * 64;
#pragma unroll
  for (int i = 0; i < 2; ++i) {
    float pv = v * es[lane + 32 * i];
    __hip_atomic_fetch_add(&ar[128 + lane + 32 * i], pv, __ATOMIC_RELAXED,
                           __HIP_MEMORY_SCOPE_AGENT);
  }
  if (lane == 0)
    __hip_atomic_fetch_add(&deg[r], v, __ATOMIC_RELAXED,
                           __HIP_MEMORY_SCOPE_AGENT);
}

// hcat[n, 0:128] = deg[n]*x[n]; hcat[n, 128:320] = agg[n]  (bf16, zero-padded rows)
__global__ __launch_bounds__(256) void build_hcat(
    const float* __restrict__ x, const float* __restrict__ deg,
    const float* __restrict__ agg, bf16_t* __restrict__ hcat, int Nrows,
    long total) {
  long idx = (long)blockIdx.x * 256 + threadIdx.x;
  if (idx >= total) return;
  int n = (int)(idx / 320);
  int c = (int)(idx % 320);
  float v = 0.f;
  if (n < Nrows) {
    if (c < 128)
      v = deg[n] * x[(size_t)n * 128 + c];  // (1 + GIN_EPS) == 1
    else
      v = agg[(size_t)n * 192 + (c - 128)];
  }
  hcat[idx] = f2bf(v);
}

// Swizzle a [K x 128] fp32 weight into WMMA-B fragment-major bf16:
// frag f = kk*8 + ntile; element (lane, j):  K = kk*32 + (lane>>4)*16 + j,
// n = ntile*16 + (lane&15). Optional per-K scale (BN fold).
template <int KSTEPS>
__global__ __launch_bounds__(256) void swizzle_B(const float* __restrict__ W,
                                                 const float* __restrict__ scale,
                                                 bf16_t* __restrict__ out) {
  int idx = (int)(blockIdx.x * 256 + threadIdx.x);
  constexpr int total = KSTEPS * 8 * 32 * 16;
  if (idx >= total) return;
  int j = idx & 15;
  int lane = (idx >> 4) & 31;
  int f = idx >> 9;
  int t = f & 7;
  int kk = f >> 3;
  int n = t * 16 + (lane & 15);
  int K = kk * 32 + (lane >> 4) * 16 + j;
  float v = W[(size_t)K * 128 + n];
  if (scale) v *= scale[K];
  out[idx] = f2bf(v);
}

// GEMM: [Npad x (KSTEPS*32)] (bf16, row-major, stride lda) @ swizzled-B -> 128 cols.
// 256 threads = 8 waves; wave w owns rows [blk*128 + w*16, +16), 8 WMMA n-tiles.
// FUSE1: +bias, ReLU, store bf16, accumulate per-block BN column sums/sumsq.
// else : +bias, store fp32 rows < validRows (final output).
template <int KSTEPS, bool FUSE1>
__global__ __launch_bounds__(256) void gemm_wmma_k(
    const bf16_t* __restrict__ A, int lda, const bf16_t* __restrict__ Bs,
    const float* __restrict__ bias, bf16_t* __restrict__ outBf,
    float* __restrict__ outF, float* __restrict__ partials, int validRows) {
  constexpr int NT = 8;
  __shared__ float colsum[128];
  __shared__ float colsq[128];
  const int lane = threadIdx.x & 31;
  const int wave = threadIdx.x >> 5;
  const int half = lane >> 4;
  const int nrow = lane & 15;
  const int mbase = blockIdx.x * 128 + wave * 16;

  if (FUSE1) {
    if (threadIdx.x < 128) {
      colsum[threadIdx.x] = 0.f;
      colsq[threadIdx.x] = 0.f;
    }
    __syncthreads();
  }

  v8f zero = {};
  v8f acc[NT];
#pragma unroll
  for (int t = 0; t < NT; ++t) acc[t] = zero;

  // A-fragment (matches swizzled B): lanes 0-15 -> K {0..15},
  // lanes 16-31 -> K {16..31}; two contiguous 16B loads per lane.
  const bf16_t* arow = A + (size_t)(mbase + nrow) * lda;
#pragma unroll
  for (int kk = 0; kk < KSTEPS; ++kk) {
    const int k0 = kk * 32 + half * 16;
    v8bf alo = *(const v8bf*)(arow + k0);
    v8bf ahi = *(const v8bf*)(arow + k0 + 8);
    v16bf afrag = __builtin_shufflevector(alo, ahi, 0, 1, 2, 3, 4, 5, 6, 7, 8,
                                          9, 10, 11, 12, 13, 14, 15);
#pragma unroll
    for (int t = 0; t < NT; ++t) {
      v16bf bfrag = *(const v16bf*)(Bs + ((size_t)(kk * NT + t) * 32 + lane) * 16);
      acc[t] = __builtin_amdgcn_wmma_f32_16x16x32_bf16(
          false, afrag, false, bfrag, (short)0, acc[t], false, false);
    }
  }

  // D layout: lane n = nrow; VGPR r holds row (r + 8*half) of the tile.
#pragma unroll
  for (int t = 0; t < NT; ++t) {
    const int n = t * 16 + nrow;
    const float bv = bias[n];
    if (FUSE1) {
      float s = 0.f, q = 0.f;
#pragma unroll
      for (int r = 0; r < 8; ++r) {
        const int gm = mbase + r + half * 8;
        float v = acc[t][r] + bv;
        v = v > 0.f ? v : 0.f;
        outBf[(size_t)gm * 128 + n] = f2bf(v);
        if (gm < validRows) {
          s += v;
          q += v * v;
        }
      }
      atomicAdd(&colsum[n], s);
      atomicAdd(&colsq[n], q);
    } else {
#pragma unroll
      for (int r = 0; r < 8; ++r) {
        const int gm = mbase + r + half * 8;
        if (gm < validRows) outF[(size_t)gm * 128 + n] = acc[t][r] + bv;
      }
    }
  }

  if (FUSE1) {
    __syncthreads();
    if (threadIdx.x < 128) {
      partials[(size_t)blockIdx.x * 256 + threadIdx.x] = colsum[threadIdx.x];
      partials[(size_t)blockIdx.x * 256 + 128 + threadIdx.x] = colsq[threadIdx.x];
    }
  }
}

// Per-feature BN: s = gamma * rsqrt(var + eps); shift = beta - mu*s.
__global__ void bn_finalize(const float* __restrict__ partials, int nblk,
                            const float* __restrict__ gamma,
                            const float* __restrict__ beta, int Nrows,
                            float* __restrict__ svec, float* __restrict__ bfold) {
  int j = threadIdx.x;
  float S = 0.f, Q = 0.f;
  for (int b = 0; b < nblk; ++b) {
    S += partials[(size_t)b * 256 + j];
    Q += partials[(size_t)b * 256 + 128 + j];
  }
  float inv = 1.f / (float)Nrows;
  float mu = S * inv;
  float var = Q * inv - mu * mu;
  float s = gamma[j] * rsqrtf(var + 1e-5f);
  svec[j] = s;
  bfold[j] = beta[j] - mu * s;
}

// bprime[n] = b2[n] + sum_k bfold[k] * W2[k,n]   (folds BN shift through GEMM2)
__global__ void fold_bias(const float* __restrict__ bfold,
                          const float* __restrict__ W2,
                          const float* __restrict__ b2,
                          float* __restrict__ bprime) {
  int n = threadIdx.x;
  float s = b2[n];
  for (int k = 0; k < 128; ++k) s += bfold[k] * W2[(size_t)k * 128 + n];
  bprime[n] = s;
}

// ---------------------------------------------------------------------------
extern "C" void kernel_launch(void* const* d_in, const int* in_sizes, int n_in,
                              void* d_out, int out_size, void* d_ws,
                              size_t ws_size, hipStream_t stream) {
  const float* x     = (const float*)d_in[0];
  const float* vals  = (const float*)d_in[1];
  const float* ea    = (const float*)d_in[2];
  const float* W1    = (const float*)d_in[3];
  const float* b1    = (const float*)d_in[4];
  const float* gamma = (const float*)d_in[5];
  const float* beta  = (const float*)d_in[6];
  const float* W2    = (const float*)d_in[7];
  const float* b2    = (const float*)d_in[8];
  const int*   row   = (const int*)d_in[9];
  const int*   col   = (const int*)d_in[10];
  (void)n_in; (void)out_size; (void)ws_size;

  const int D = 128;
  int N = in_sizes[0] / D;
  int E = in_sizes[1];
  int nblk = (N + 127) / 128;
  int Npad = nblk * 128;

  // Workspace carve (~170 MB): fp32 accumulators + bf16 staging buffers.
  char* ws = (char*)d_ws;
  size_t off = 0;
  auto carve = [&](size_t bytes) -> void* {
    void* p = ws + off;
    off += (bytes + 255) & ~(size_t)255;
    return p;
  };
  float*  deg      = (float*)carve((size_t)Npad * 4);
  float*  agg      = (float*)carve((size_t)Npad * 192 * 4);
  bf16_t* hcat     = (bf16_t*)carve((size_t)Npad * 320 * 2);
  bf16_t* h1       = (bf16_t*)carve((size_t)Npad * 128 * 2);
  bf16_t* W1s      = (bf16_t*)carve((size_t)320 * 128 * 2);
  bf16_t* W2s      = (bf16_t*)carve((size_t)128 * 128 * 2);
  float*  partials = (float*)carve((size_t)nblk * 256 * 4);
  float*  svec     = (float*)carve(128 * 4);
  float*  bfold    = (float*)carve(128 * 4);
  float*  bprime   = (float*)carve(128 * 4);

  // 1. zero scatter accumulators (must re-zero every call)
  {
    size_t n1 = (size_t)Npad;
    zero_f32<<<(unsigned)((n1 + 255) / 256), 256, 0, stream>>>(deg, n1);
    size_t n2 = (size_t)Npad * 192;
    zero_f32<<<(unsigned)((n2 + 255) / 256), 256, 0, stream>>>(agg, n2);
  }
  // 2. edge gather/scale/scatter (one wave per edge)
  edge_scatter<<<(E + 7) / 8, 256, 0, stream>>>(x, vals, ea, row, col, deg, agg, E);
  // 3. bf16 concat [Npad x 320]
  {
    long total = (long)Npad * 320;
    build_hcat<<<(unsigned)((total + 255) / 256), 256, 0, stream>>>(x, deg, agg,
                                                                    hcat, N, total);
  }
  // 4. W1 -> fragment-major bf16
  swizzle_B<10><<<(10 * 8 * 32 * 16 + 255) / 256, 256, 0, stream>>>(W1, nullptr, W1s);
  // 5. GEMM1 + bias + ReLU + BN partial sums
  gemm_wmma_k<10, true><<<nblk, 256, 0, stream>>>(hcat, 320, W1s, b1, h1,
                                                  nullptr, partials, N);
  // 6. BN stats -> per-feature scale/shift
  bn_finalize<<<1, 128, 0, stream>>>(partials, nblk, gamma, beta, N, svec, bfold);
  // 7. fold BN shift through W2 into bias
  fold_bias<<<1, 128, 0, stream>>>(bfold, W2, b2, bprime);
  // 8. W2 scaled by BN -> fragment-major bf16
  swizzle_B<4><<<(4 * 8 * 32 * 16 + 255) / 256, 256, 0, stream>>>(W2, svec, W2s);
  // 9. GEMM2 + folded bias -> fp32 output
  gemm_wmma_k<4, false><<<nblk, 256, 0, stream>>>(h1, 128, W2s, bprime, nullptr,
                                                  (float*)d_out, nullptr, N);
}
